// RNN_72834055406304
// MI455X (gfx1250) — matrix operand order repaired and checked by reading, not compiled, not run on hardware
//
#include <hip/hip_runtime.h>

#define NB 32
#define TT 600
#define NIN 16
#define NH 2048
#define NO 16
#define GAINF 3.2f          // 1.6 / 0.5
#define DT_TAU 0.1f         // DT / TAU
#define DTF 0.001f
#define BETAF 20.0f
#define INV_TD 5.0f         // 1 / TAU_DEPR
#define INV_TF 0.6666666667f// 1 / TAU_FACIL
#define PSTR 2056           // padded LDS row stride (bf16 elems) to avoid bank conflicts
#define GBLK 16             // blocks in step kernel; each owns 128 columns + 2 batches for z

typedef __attribute__((ext_vector_type(16))) __bf16 v16bf;
typedef __attribute__((ext_vector_type(8)))  float v8f;

__device__ __forceinline__ unsigned short f2bf(float f) {
    unsigned u = __float_as_uint(f);
    u += 0x7FFFu + ((u >> 16) & 1u);       // round-to-nearest-even
    return (unsigned short)(u >> 16);
}

// One-time: pack (W_hh*mask)^T into WMMA B-fragment tile layout, bf16.
// Slot idx = (kc<<12)|(nt<<5)|lane ; element hv: K = kc*32+16*(lane>>4)+hv, N = nt*16+(lane&15)
__global__ void pack_weights(const float* __restrict__ W, const float* __restrict__ M,
                             unsigned short* __restrict__ wpk) {
    int idx  = blockIdx.x * 256 + threadIdx.x;   // 64*128*32 = 262144 lane-slots
    int lane = idx & 31;
    int nt   = (idx >> 5) & 127;
    int kc   = idx >> 12;
    int n    = nt * 16 + (lane & 15);
    int kb   = kc * 32 + ((lane >> 4) << 4);
    const float* wr = W + n * NH + kb;           // contiguous 16 floats (row n of W_hh)
    const float* mr = M + n * NH + kb;
    unsigned short o[16];
    #pragma unroll
    for (int hv = 0; hv < 16; ++hv) o[hv] = f2bf(wr[hv] * mr[hv]);
    uint4* dst = (uint4*)(wpk + (idx << 4));
    dst[0] = ((uint4*)o)[0];
    dst[1] = ((uint4*)o)[1];
}

// One-time: state init + bootstrap pre(0) = tanh(h0)*r0*u0*scale (bf16, row-major)
__global__ void init_state(const float* __restrict__ h0, const float* __restrict__ r0,
                           const float* __restrict__ u0, const float* __restrict__ psyn,
                           float* __restrict__ h, float* __restrict__ htr,
                           float* __restrict__ r, float* __restrict__ u,
                           unsigned short* __restrict__ pg0) {
    int e = blockIdx.x * 256 + threadIdx.x;      // 65536
    int j = e & (NH - 1);
    float hv = h0[e];
    float th = tanhf(hv);
    h[e] = hv; htr[e] = th; r[e] = r0[e]; u[e] = u0[e];
    pg0[e] = f2bf(th * r0[e] * u0[e] * psyn[j] * GAINF);
}

// Fused per-timestep kernel. 16 blocks x 256 threads (8 waves).
// Block bl: readout z for batches {2bl, 2bl+1}, owns columns [bl*128, bl*128+128).
// pre(t) arrives as a bf16 global image (written by step t-1's epilogue) and is
// async-copied into LDS (ASYNCcnt) overlapped with the z-readout and r/u update.
__global__ void __launch_bounds__(256) rnn_step(
    const float* __restrict__ x, const float* __restrict__ p_rel,
    const float* __restrict__ W_ih, const float* __restrict__ presyn,
    const float* __restrict__ W_hz, const unsigned short* __restrict__ wpk,
    const float* __restrict__ htr_in, float* __restrict__ htr_out,
    float* __restrict__ r_state, float* __restrict__ u_state,
    const unsigned short* __restrict__ pg_in, unsigned short* __restrict__ pg_out,
    float* __restrict__ h_state,
    float* __restrict__ h_all, float* __restrict__ r_all,
    float* __restrict__ u_all, float* __restrict__ z_all, int t)
{
    __shared__ __align__(16) unsigned short preS[NB * PSTR];  // bf16 pre, padded rows
    __shared__ float ruS[NB * 128];                           // r_new*u_new*scale for owned slice
    __shared__ float xs[NB * NIN];
    __shared__ float zs[256];
    const int tid = threadIdx.x;
    const int bl  = blockIdx.x;

    // ---- kick off async copy pre(t): global bf16 -> padded LDS (ASYNCcnt) ----
    if (t < TT) {
        unsigned ldsbase = (unsigned)(size_t)&preS[0];
        unsigned long long gbase = (unsigned long long)(size_t)pg_in;
        #pragma unroll 4
        for (int it = 0; it < 32; ++it) {
            int e   = tid + it * 256;          // 8192 x 16-byte chunks
            int b   = e >> 8;                  // 256 chunks per 4096B row
            int c16 = e & 255;
            unsigned lds          = ldsbase + (unsigned)(b * (PSTR * 2) + c16 * 16);
            unsigned long long ga = gbase + (unsigned long long)(b * 4096 + c16 * 16);
            asm volatile("global_load_async_to_lds_b128 %0, %1, off"
                         :: "v"(lds), "v"(ga) : "memory");
        }
    }

    // ---- readout z(t-1) = tanh(h(t-1)) @ W_hz^T  (uses htr_in); 2 batches per block ----
    if (t > 0) {
        int b  = (bl << 1) + (tid >> 7);       // 2 batches
        int o  = tid & 15;
        int ch = (tid >> 4) & 7;               // 8 K-chunks of 256
        const float* hr = htr_in + b * NH + ch * 256;
        const float* wz = W_hz   + o * NH + ch * 256;
        float acc = 0.f;
        #pragma unroll 4
        for (int k = 0; k < 256; ++k) acc += hr[k] * wz[k];
        zs[tid] = acc;
        __syncthreads();
        if (tid < 32) {
            int bb = tid >> 4, oo = tid & 15;
            float s = 0.f;
            #pragma unroll
            for (int c = 0; c < 8; ++c) s += zs[bb * 128 + c * 16 + oo];
            z_all[(((bl << 1) + bb) * TT + (t - 1)) * NO + oo] = s;
        }
    }
    if (t >= TT) return;     // tail launch: readout only (implicit wait-idle at endpgm)

    // ---- stage x_t into LDS ----
    for (int e = tid; e < NB * NIN; e += 256) {
        int b = e >> 4, q = e & 15;
        xs[e] = x[(b * TT + t) * NIN + q];
    }
    // ---- STP state update for owned column slice (owner-exclusive, in place) ----
    const int j0 = bl * 128;
    for (int e = tid; e < NB * 128; e += 256) {
        int b = e >> 7, j = j0 + (e & 127);
        float rv = r_state[b * NH + j], uv = u_state[b * NH + j], hv = htr_in[b * NH + j];
        float pj = p_rel[j];
        float rn = rv + ((1.f - rv) * INV_TD - BETAF * uv * rv * hv) * DTF;
        float un = uv + ((pj - uv) * INV_TF + BETAF * pj * (1.f - uv) * hv) * DTF;
        r_state[b * NH + j] = rn;  u_state[b * NH + j] = un;
        r_all[(b * TT + t) * NH + j] = rn;
        u_all[(b * TT + t) * NH + j] = un;
        ruS[e] = rn * un * presyn[j] * GAINF;   // for pre(t+1) in epilogue
    }
    asm volatile("s_wait_asynccnt 0x0" ::: "memory");
    __syncthreads();

    // ---- GEMM: out[32 x 128-slice] = pre @ WmT  via v_wmma_f32_16x16x32_bf16 ----
    // One N-tile per wave; B fragment loaded once, used by both M-tiles.
    const int wave = tid >> 5, lane = tid & 31;
    const int ntg  = bl * 8 + wave;            // global 16-col tile index (0..127)
    const int n0   = ntg << 4;
    const int mA   = lane & 15;                // A-frag row, M-tile 0
    const int mB   = 16 + (lane & 15);         // A-frag row, M-tile 1
    const int koff = (lane >> 4) << 4;         // A: K offset 0 or 16
    union AB { uint4 q[2]; v16bf v; };
    v8f acc0 = {}, acc1 = {};
    #pragma unroll 2
    for (int kc = 0; kc < 64; ++kc) {
        const int k0 = kc << 5;
        AB a0, a1, bfr;
        const uint4* bp = (const uint4*)(wpk + (((kc * 128 + ntg) * 32 + lane) << 4));
        bfr.q[0] = bp[0];                                           // global_load_b128
        bfr.q[1] = bp[1];
        if (kc < 63) __builtin_prefetch(bp + 8192, 0, 3);           // next K-chunk -> near cache
        a0.q[0] = *(const uint4*)&preS[mA * PSTR + k0 + koff];      // ds_load_b128
        a0.q[1] = *(const uint4*)&preS[mA * PSTR + k0 + 8 + koff];
        a1.q[0] = *(const uint4*)&preS[mB * PSTR + k0 + koff];
        a1.q[1] = *(const uint4*)&preS[mB * PSTR + k0 + 8 + koff];
        acc0 = __builtin_amdgcn_wmma_f32_16x16x32_bf16(
                   false, a0.v, false, bfr.v, (short)0, acc0, false, false);
        acc1 = __builtin_amdgcn_wmma_f32_16x16x32_bf16(
                   false, a1.v, false, bfr.v, (short)0, acc1, false, false);
    }

    // ---- epilogue: h update + tanh + outputs + pre(t+1) (C/D: VGPR c -> M=c+8*(lane>=16)) ----
    const int i  = n0 + (lane & 15);
    const int ii = i - j0;                     // 0..127 within owned slice
    float wih[16];
    #pragma unroll
    for (int q = 0; q < 16; ++q) wih[q] = W_ih[i * NIN + q];
    const int bbase = (lane >> 4) << 3;
    #pragma unroll
    for (int c = 0; c < 8; ++c) {
        int b0 = bbase + c;                    // M-tile 0 row
        int b1 = b0 + 16;                      // M-tile 1 row
        float xw0 = 0.f, xw1 = 0.f;
        #pragma unroll
        for (int q = 0; q < 16; ++q) {
            xw0 += xs[b0 * NIN + q] * wih[q];
            xw1 += xs[b1 * NIN + q] * wih[q];
        }
        float h0v = h_state[b0 * NH + i];
        float h1v = h_state[b1 * NH + i];
        float hn0 = h0v + DT_TAU * (-h0v + xw0 + acc0[c]);
        float hn1 = h1v + DT_TAU * (-h1v + xw1 + acc1[c]);
        float th0 = tanhf(hn0);
        float th1 = tanhf(hn1);
        h_state[b0 * NH + i] = hn0;
        h_state[b1 * NH + i] = hn1;
        h_all[(b0 * TT + t) * NH + i] = hn0;
        h_all[(b1 * TT + t) * NH + i] = hn1;
        htr_out[b0 * NH + i] = th0;
        htr_out[b1 * NH + i] = th1;
        pg_out[b0 * NH + i] = f2bf(th0 * ruS[b0 * 128 + ii]);   // pre(t+1), bf16
        pg_out[b1 * NH + i] = f2bf(th1 * ruS[b1 * 128 + ii]);
    }
}

extern "C" void kernel_launch(void* const* d_in, const int* in_sizes, int n_in,
                              void* d_out, int out_size, void* d_ws, size_t ws_size,
                              hipStream_t stream) {
    const float* x     = (const float*)d_in[0];
    const float* h0    = (const float*)d_in[1];
    const float* r0    = (const float*)d_in[2];
    const float* u0    = (const float*)d_in[3];
    const float* prel  = (const float*)d_in[4];
    const float* Wih   = (const float*)d_in[5];
    const float* Whh   = (const float*)d_in[6];
    const float* Wmask = (const float*)d_in[7];
    const float* psyn  = (const float*)d_in[8];
    const float* Whz   = (const float*)d_in[9];

    char* ws = (char*)d_ws;
    unsigned short* wpk = (unsigned short*)ws;  ws += (size_t)NH * NH * 2;   // 8 MB packed bf16 weights
    float* htr[2];
    htr[0] = (float*)ws; ws += (size_t)NB * NH * 4;
    htr[1] = (float*)ws; ws += (size_t)NB * NH * 4;
    float* rst    = (float*)ws; ws += (size_t)NB * NH * 4;
    float* ust    = (float*)ws; ws += (size_t)NB * NH * 4;
    float* hstate = (float*)ws; ws += (size_t)NB * NH * 4;
    unsigned short* pg[2];
    pg[0] = (unsigned short*)ws; ws += (size_t)NB * NH * 2;                  // 128 KB bf16 pre image
    pg[1] = (unsigned short*)ws; ws += (size_t)NB * NH * 2;

    float* out   = (float*)d_out;
    float* h_all = out;
    float* r_all = h_all + (size_t)NB * TT * NH;
    float* u_all = r_all + (size_t)NB * TT * NH;
    float* z_all = u_all + (size_t)NB * TT * NH;

    pack_weights<<<1024, 256, 0, stream>>>(Whh, Wmask, wpk);
    init_state<<<256, 256, 0, stream>>>(h0, r0, u0, psyn, hstate, htr[0], rst, ust, pg[0]);
    for (int t = 0; t <= TT; ++t) {
        int pi = t & 1, po = pi ^ 1;
        rnn_step<<<GBLK, 256, 0, stream>>>(x, prel, Wih, psyn, Whz, wpk,
                                           htr[pi], htr[po], rst, ust,
                                           pg[pi], pg[po],
                                           hstate, h_all, r_all, u_all, z_all, t);
    }
}
